// HeteroGATConv_46351287058832
// MI455X (gfx1250) — matrix operand order repaired and checked by reading, not compiled, not run on hardware
//
#include <hip/hip_runtime.h>

// ---------------------------------------------------------------------------
// HeteroGATConv for MI455X (gfx1250, wave32).
//   Phase 0: convert x to bf16; swizzle W into the exact WMMA B-operand
//            register image (per lane, per 16-col tile, per K-step); build
//            per-head attention vectors wvec[h] = W[:,h*64:(h+1)*64] @ a[h].
//   Per relation (sequential on stream, workspace reused):
//     1) XS = Xsrc_bf16 @ W_bf16 via v_wmma_f32_16x16x32_bf16 (f32 accum);
//        the block's 16KB B-panel is staged in LDS once via
//        global_load_async_to_lds_b128 (ASYNCcnt) and read as ds_load_b128.
//        The LDS address operand is derived from the shared object through an
//        addrspace(3) cast so the object escapes into the asm (correct
//        address AND forces the compiler to treat LDS as written).
//     2) al_src/al_dst = x . wvec  (128-long dots, wvec staged in LDS)
//     3) segment softmax: atomicMax (monotone-uint float encoding), then
//        atomicAdd of exp
//     4) message scatter: one wave per edge, head-mean + hetero-mean folded
//        into the 0.125 factor so atomics hit only [Nd,64] (4x less traffic)
// ---------------------------------------------------------------------------

typedef __attribute__((ext_vector_type(16))) __bf16 bf16x16;
typedef __attribute__((ext_vector_type(8)))  float  floatx8;
typedef __attribute__((address_space(3))) unsigned char lds_byte_t;

union AV { bf16x16 v; unsigned int u[8]; };
union BV { bf16x16 v; uint4 q[2]; };

#define NEG_SLOPE 0.2f

__device__ __forceinline__ unsigned short f2bf(float f) {
  unsigned u = __float_as_uint(f);
  unsigned rnd = 0x7FFFu + ((u >> 16) & 1u);   // round-to-nearest-even
  return (unsigned short)((u + rnd) >> 16);
}
__device__ __forceinline__ unsigned fenc(float f) {  // monotone float->uint
  int b = __float_as_int(f);
  return (b >= 0) ? ((unsigned)b | 0x80000000u) : ~(unsigned)b;
}
__device__ __forceinline__ float fdec(unsigned u) {
  int b = (u & 0x80000000u) ? (int)(u & 0x7FFFFFFFu) : ~(int)u;
  return __int_as_float(b);
}
__device__ __forceinline__ float lrelu(float x) {
  return x > 0.f ? x : NEG_SLOPE * x;
}

// ---- phase 0 kernels -------------------------------------------------------

__global__ void k_cvt_bf16(const float* __restrict__ x,
                           unsigned short* __restrict__ y, long n) {
  long i = (long)blockIdx.x * blockDim.x + threadIdx.x;
  if (i < n) y[i] = f2bf(x[i]);
}

// Wsw[r][tile][ks][lane][j] = pack(W[r][K][col], W[r][K+1][col]) with
//   col = tile*16 + (lane&15), K = ks*32 + ((lane&16)?16:0) + 2j
// i.e. the per-lane B-operand dwords in contiguous order K = 16h + s (A uses the same).
__global__ void k_swizzle_w(const float* __restrict__ W,
                            unsigned int* __restrict__ Wsw) {
  int i = blockIdx.x * blockDim.x + threadIdx.x;   // 4*16*4*32*8 = 65536
  if (i >= 65536) return;
  int j    = i & 7;
  int lane = (i >> 3) & 31;
  int ks   = (i >> 8) & 3;
  int t    = (i >> 10) & 15;
  int r    = (i >> 14) & 3;
  int col = t * 16 + (lane & 15);
  int K   = ks * 32 + ((lane & 16) ? 16 : 0) + 2 * j;
  const float* base = W + (size_t)r * 128 * 256 + (size_t)K * 256 + col;
  unsigned lo = f2bf(base[0]);
  unsigned hi = f2bf(base[256]);
  Wsw[i] = lo | (hi << 16);
}

// wvec[r][h][k] = sum_c W[r][k][h*64+c] * att[r][h][c]
__global__ void k_make_wvec(const float* __restrict__ W,
                            const float* __restrict__ att,
                            float* __restrict__ wvec) {
  int i = blockIdx.x * blockDim.x + threadIdx.x;   // 4*4*128
  if (i >= 2048) return;
  int r = i >> 9, h = (i >> 7) & 3, k = i & 127;
  const float* Wr = W + (size_t)r * 128 * 256 + (size_t)k * 256 + h * 64;
  const float* ar = att + r * 256 + h * 64;
  float acc = 0.f;
#pragma unroll 8
  for (int c = 0; c < 64; ++c) acc += Wr[c] * ar[c];
  wvec[((size_t)r * 4 + h) * 128 + k] = acc;
}

// out_agent init = 0.5*(b1+b3); out_track init = 0.5*(b0+b2)
__global__ void k_init_out(const float* __restrict__ bias,
                           float* __restrict__ out, long total, long agentElems) {
  long i = (long)blockIdx.x * blockDim.x + threadIdx.x;
  if (i >= total) return;
  int c = (int)(i & 63);
  out[i] = (i < agentElems) ? 0.5f * (bias[64 + c] + bias[192 + c])
                            : 0.5f * (bias[c] + bias[128 + c]);
}

__global__ void k_init_ms(unsigned* __restrict__ m, float* __restrict__ s, int n) {
  int i = blockIdx.x * blockDim.x + threadIdx.x;
  if (i < n) { m[i] = 0u; s[i] = 0.f; }   // 0 < fenc(x) for all finite x
}

// ---- GEMM: XS[Ns,256] = X16[Ns,128] @ W_bf16[128,256], f32 accum -----------
// Block = 256 threads = 8 waves; wave w -> rows [bx*128+16w, +16), cols
// [by*64, +64) as four 16x16 WMMA accumulators sharing one A operand.
// The block's 16KB B-panel (4 tiles x 4 ksteps x 32 lanes x 32B) is staged
// once into LDS with async copies, then read with ds_load_b128.
__global__ __launch_bounds__(256) void k_gemm_xw(
    const unsigned short* __restrict__ X16,
    const unsigned int* __restrict__ Wsw,   // this relation: [16][4][32][8] dw
    float* __restrict__ XS, int Ns) {
  __shared__ alignas(16) unsigned int bsh[4096];   // 16KB B-panel

  // True LDS byte offset of bsh: AS3 pointers are 32-bit LDS offsets.
  // Passing it into the asm also escapes bsh, so the "memory" clobber
  // legitimately may write it (prevents undef-folding of the LDS reads).
  const unsigned ldsBase =
      (unsigned)(unsigned long long)(lds_byte_t*)(void*)bsh;

  // --- async-stage B panel: 256 threads x 4 chunks of 16B -------------------
  {
    const unsigned long long gbase =
        (unsigned long long)(const void*)(Wsw + (size_t)blockIdx.y * 4096);
    const unsigned tid = threadIdx.x;
#pragma unroll
    for (int it = 0; it < 4; ++it) {
      unsigned idx = tid + 256u * it;            // 16B-chunk index 0..1023
      unsigned lds = ldsBase + idx * 16u;
      unsigned long long ga = gbase + (unsigned long long)idx * 16u;
      asm volatile("global_load_async_to_lds_b128 %0, %1, off"
                   :: "v"(lds), "v"(ga) : "memory");
    }
    asm volatile("s_wait_asynccnt 0x0" ::: "memory");
  }
  __syncthreads();   // all threads reach this before any wave-level early-out

  const int lane = threadIdx.x & 31;
  const int wave = threadIdx.x >> 5;
  const int tileM = blockIdx.x * 128 + wave * 16;
  if (tileM >= Ns) return;                 // Ns % 16 == 0 -> whole-tile check
  const int colBase = blockIdx.y * 64;
  const int n = lane & 15;
  const bool hi = lane >= 16;

  floatx8 zero = {0.f, 0.f, 0.f, 0.f, 0.f, 0.f, 0.f, 0.f};
  floatx8 acc[4] = {zero, zero, zero, zero};

  const int row = tileM + n;               // A: M = lane & 15 (both halves)
  const unsigned int* Arow = (const unsigned int*)(X16 + (size_t)row * 128);
  const int klA = hi ? 16 : 0;             // A K-split matches B: K = 16h + s
  const uint4* bv = (const uint4*)bsh;

#pragma unroll
  for (int ks = 0; ks < 4; ++ks) {
    const int k0 = ks * 32;
    if (ks < 3) __builtin_prefetch(Arow + ((k0 + 32) >> 1), 0, 1);
    AV a;
#pragma unroll
    for (int j = 0; j < 4; ++j) {
      a.u[j]     = Arow[(k0 + klA + 2 * j) >> 1];        // K = k0+klA+{0..7}
      a.u[4 + j] = Arow[(k0 + klA + 8 + 2 * j) >> 1];    // K = k0+klA+{8..15}
    }
#pragma unroll
    for (int t = 0; t < 4; ++t) {
      BV b;
      const int bidx = ((t * 4 + ks) * 32 + lane) * 2;   // uint4 index in LDS
      b.q[0] = bv[bidx];
      b.q[1] = bv[bidx + 1];
      acc[t] = __builtin_amdgcn_wmma_f32_16x16x32_bf16(
          false, a.v, false, b.v, (short)0, acc[t], false, false);
    }
  }
#pragma unroll
  for (int t = 0; t < 4; ++t) {
    const int col = colBase + t * 16 + n;
#pragma unroll
    for (int j = 0; j < 8; ++j) {          // C/D: VGPR j -> M = j (+8 hi half)
      const int mrow = tileM + (hi ? 8 + j : j);
      XS[(size_t)mrow * 256 + col] = acc[t][j];
    }
  }
}

// ---- attention logits: al[i][h] = x[i,:] . wvec[h,:] -----------------------
__global__ __launch_bounds__(256) void k_node_al(
    const float* __restrict__ x, const float* __restrict__ wvec,
    float* __restrict__ al, int N) {
  __shared__ float wv[512];
  for (int t = threadIdx.x; t < 512; t += blockDim.x) wv[t] = wvec[t];
  __syncthreads();
  int i = blockIdx.x * blockDim.x + threadIdx.x;
  if (i >= N) return;
  const float4* xr = (const float4*)(x + (size_t)i * 128);
  float a[4] = {0.f, 0.f, 0.f, 0.f};
#pragma unroll 4
  for (int q = 0; q < 32; ++q) {
    float4 v = xr[q];
    int k = q * 4;
#pragma unroll
    for (int h = 0; h < 4; ++h) {
      const float* w = wv + h * 128 + k;
      a[h] += v.x * w[0] + v.y * w[1] + v.z * w[2] + v.w * w[3];
    }
  }
  float4 r4 = {a[0], a[1], a[2], a[3]};
  *(float4*)(al + (size_t)i * 4) = r4;
}

// ---- softmax pass 1: per-dst max via uint-encoded atomicMax ----------------
__global__ void k_edge_max(const int* __restrict__ es, const int* __restrict__ ed,
                           const float* __restrict__ als, const float* __restrict__ ald,
                           unsigned* __restrict__ m, int E) {
  int e = blockIdx.x * blockDim.x + threadIdx.x;
  if (e >= E) return;
  int s = es[e], d = ed[e];
  float4 s4 = *(const float4*)(als + (size_t)s * 4);
  float4 d4 = *(const float4*)(ald + (size_t)d * 4);
  unsigned* mp = m + (size_t)d * 4;
  atomicMax(mp + 0, fenc(lrelu(s4.x + d4.x)));
  atomicMax(mp + 1, fenc(lrelu(s4.y + d4.y)));
  atomicMax(mp + 2, fenc(lrelu(s4.z + d4.z)));
  atomicMax(mp + 3, fenc(lrelu(s4.w + d4.w)));
}

// ---- softmax pass 2: per-dst sum of exp ------------------------------------
__global__ void k_edge_sum(const int* __restrict__ es, const int* __restrict__ ed,
                           const float* __restrict__ als, const float* __restrict__ ald,
                           const unsigned* __restrict__ m, float* __restrict__ ssum,
                           int E) {
  int e = blockIdx.x * blockDim.x + threadIdx.x;
  if (e >= E) return;
  int s = es[e], d = ed[e];
  float4 s4 = *(const float4*)(als + (size_t)s * 4);
  float4 d4 = *(const float4*)(ald + (size_t)d * 4);
  const unsigned* mp = m + (size_t)d * 4;
  float* sp = ssum + (size_t)d * 4;
  atomicAdd(sp + 0, __expf(lrelu(s4.x + d4.x) - fdec(mp[0])));
  atomicAdd(sp + 1, __expf(lrelu(s4.y + d4.y) - fdec(mp[1])));
  atomicAdd(sp + 2, __expf(lrelu(s4.z + d4.z) - fdec(mp[2])));
  atomicAdd(sp + 3, __expf(lrelu(s4.w + d4.w) - fdec(mp[3])));
}

// ---- pass 3: one wave per edge; out[dst,c] += 0.125 * sum_h a_h*XS[src,h,c]
__global__ __launch_bounds__(256) void k_edge_scatter(
    const int* __restrict__ es, const int* __restrict__ ed,
    const float* __restrict__ als, const float* __restrict__ ald,
    const unsigned* __restrict__ m, const float* __restrict__ ssum,
    const float* __restrict__ XS, float* __restrict__ out, int E) {
  int e = blockIdx.x * 8 + (threadIdx.x >> 5);
  if (e >= E) return;
  int lane = threadIdx.x & 31;
  int s = es[e], d = ed[e];
  const float* xr = XS + (size_t)s * 256;
  __builtin_prefetch(xr + lane, 0, 1);
  float4 s4 = *(const float4*)(als + (size_t)s * 4);
  float4 d4 = *(const float4*)(ald + (size_t)d * 4);
  const unsigned* mp = m + (size_t)d * 4;
  const float* sp = ssum + (size_t)d * 4;
  float a0 = __expf(lrelu(s4.x + d4.x) - fdec(mp[0])) / (sp[0] + 1e-16f);
  float a1 = __expf(lrelu(s4.y + d4.y) - fdec(mp[1])) / (sp[1] + 1e-16f);
  float a2 = __expf(lrelu(s4.z + d4.z) - fdec(mp[2])) / (sp[2] + 1e-16f);
  float a3 = __expf(lrelu(s4.w + d4.w) - fdec(mp[3])) / (sp[3] + 1e-16f);
  int c0 = lane, c1 = lane + 32;
  float v0 = a0 * xr[c0] + a1 * xr[64 + c0] + a2 * xr[128 + c0] + a3 * xr[192 + c0];
  float v1 = a0 * xr[c1] + a1 * xr[64 + c1] + a2 * xr[128 + c1] + a3 * xr[192 + c1];
  float* op = out + (size_t)d * 64;
  atomicAdd(op + c0, 0.125f * v0);   // 0.5 (hetero mean) * 0.25 (head mean)
  atomicAdd(op + c1, 0.125f * v1);
}

// ---------------------------------------------------------------------------

extern "C" void kernel_launch(void* const* d_in, const int* in_sizes, int n_in,
                              void* d_out, int out_size, void* d_ws, size_t ws_size,
                              hipStream_t stream) {
  constexpr int NA = 50000, NT = 200000, E = 500000;

  const float* x_agent = (const float*)d_in[0];
  const float* x_track = (const float*)d_in[1];
  const float* W       = (const float*)d_in[2];
  const float* att_src = (const float*)d_in[3];
  const float* att_dst = (const float*)d_in[4];
  const float* bias    = (const float*)d_in[5];
  const int* esrc[4] = {(const int*)d_in[6],  (const int*)d_in[8],
                        (const int*)d_in[10], (const int*)d_in[12]};
  const int* edst[4] = {(const int*)d_in[7],  (const int*)d_in[9],
                        (const int*)d_in[11], (const int*)d_in[13]};
  float* out = (float*)d_out;

  // workspace carve-up (256B aligned), ~282 MB total
  size_t off = 0;
  auto carve = [&](size_t bytes) {
    size_t o = off; off = (off + bytes + 255) & ~(size_t)255; return o;
  };
  char* ws = (char*)d_ws;
  unsigned short* Xa16 = (unsigned short*)(ws + carve((size_t)NA * 128 * 2));
  unsigned short* Xt16 = (unsigned short*)(ws + carve((size_t)NT * 128 * 2));
  unsigned int*   Wsw  = (unsigned int*)  (ws + carve((size_t)65536 * 4));
  float* wsvec = (float*)(ws + carve(2048 * 4));
  float* wdvec = (float*)(ws + carve(2048 * 4));
  float* alS   = (float*)(ws + carve((size_t)NT * 4 * 4));
  float* alD   = (float*)(ws + carve((size_t)NT * 4 * 4));
  unsigned* mb = (unsigned*)(ws + carve((size_t)NT * 4 * 4));
  float* sb    = (float*)(ws + carve((size_t)NT * 4 * 4));
  float* XS    = (float*)(ws + carve((size_t)NT * 256 * 4));
  (void)ws_size;

  // phase 0
  {
    long na = (long)NA * 128, nt = (long)NT * 128;
    k_cvt_bf16<<<(int)((na + 255) / 256), 256, 0, stream>>>(x_agent, Xa16, na);
    k_cvt_bf16<<<(int)((nt + 255) / 256), 256, 0, stream>>>(x_track, Xt16, nt);
    k_swizzle_w<<<65536 / 256, 256, 0, stream>>>(W, Wsw);
    k_make_wvec<<<8, 256, 0, stream>>>(W, att_src, wsvec);
    k_make_wvec<<<8, 256, 0, stream>>>(W, att_dst, wdvec);
    long total = (long)(NA + NT) * 64;
    k_init_out<<<(int)((total + 255) / 256), 256, 0, stream>>>(
        bias, out, total, (long)NA * 64);
  }

  // per-relation pipeline (stream-ordered; buffers reused)
  for (int r = 0; r < 4; ++r) {
    const bool trkSrc = (r & 1);               // r0,2: agent->track; r1,3: track->agent
    const int Ns = trkSrc ? NT : NA;
    const int Nd = trkSrc ? NA : NT;
    const unsigned short* Xs16 = trkSrc ? Xt16 : Xa16;
    const float* xsrc = trkSrc ? x_track : x_agent;
    const float* xdst = trkSrc ? x_agent : x_track;
    float* osec = trkSrc ? out : out + (size_t)NA * 64;

    dim3 gg((Ns + 127) / 128, 4, 1);
    k_gemm_xw<<<gg, 256, 0, stream>>>(Xs16, Wsw + (size_t)r * 16384, XS, Ns);
    k_node_al<<<(Ns + 255) / 256, 256, 0, stream>>>(xsrc, wsvec + r * 512, alS, Ns);
    k_node_al<<<(Nd + 255) / 256, 256, 0, stream>>>(xdst, wdvec + r * 512, alD, Nd);
    k_init_ms<<<(Nd * 4 + 255) / 256, 256, 0, stream>>>(mb, sb, Nd * 4);
    k_edge_max<<<(E + 255) / 256, 256, 0, stream>>>(esrc[r], edst[r], alS, alD, mb, E);
    k_edge_sum<<<(E + 255) / 256, 256, 0, stream>>>(esrc[r], edst[r], alS, alD, mb, sb, E);
    k_edge_scatter<<<(E + 7) / 8, 256, 0, stream>>>(esrc[r], edst[r], alS, alD,
                                                    mb, sb, XS, osec, E);
  }
}
